// AdaptiveBoundaryLoss_5918464934040
// MI455X (gfx1250) — matrix operand order repaired and checked by reading, not compiled, not run on hardware
//
#include <hip/hip_runtime.h>

#define NUM_LABELS 150
#define FEAT 768
#define NB 1500
#define BATCH 256
#define NTRI 294528  // 768*767/2

typedef unsigned short u16;
typedef __attribute__((ext_vector_type(16))) __bf16 v16bf;
typedef __attribute__((ext_vector_type(8)))  float  v8f;
typedef __attribute__((ext_vector_type(8)))  u16    v8u;
typedef __attribute__((ext_vector_type(16))) u16    v16u;

__device__ __forceinline__ u16 f2bf(float f) {
    unsigned int u = __builtin_bit_cast(unsigned int, f);
    unsigned int r = u + 0x7FFFu + ((u >> 16) & 1u);   // round-to-nearest-even
    if ((u & 0x7F800000u) == 0x7F800000u) r = u;       // keep Inf/NaN bits
    return (u16)(r >> 16);
}

__device__ __forceinline__ v16bf ldfrag(const u16* p0, const u16* p1) {
    v8u a = *(const v8u*)p0;
    v8u b = *(const v8u*)p1;
    v16u u = __builtin_shufflevector(a, b, 0,1,2,3,4,5,6,7,8,9,10,11,12,13,14,15);
    return __builtin_bit_cast(v16bf, u);
}

// ---- convert centroids + ood to bf16 (row-major: already K-major per B column) ----
__global__ void k_convert(const float* __restrict__ cent, const float* __restrict__ ood,
                          u16* __restrict__ centb, u16* __restrict__ oodb) {
    int row = blockIdx.x;
    const float* src; u16* dst;
    if (row < NB) { src = cent + (size_t)row * FEAT; dst = centb + (size_t)row * FEAT; }
    else { int b = row - NB; src = ood + (size_t)b * FEAT; dst = oodb + (size_t)b * FEAT; }
    for (int j = threadIdx.x; j < FEAT; j += blockDim.x) dst[j] = f2bf(src[j]);
}

// ---- materialize R[k] (768x768) in bf16: lower<-L, upper<-U, diag<-Dd ----
__global__ void k_build_R(const float* __restrict__ L, const float* __restrict__ U,
                          const float* __restrict__ Dd, u16* __restrict__ R) {
    int id = blockIdx.x;
    int k = id / FEAT, i = id % FEAT;
    size_t base = ((size_t)k * FEAT + i) * FEAT;
    int tri_i = i * (i - 1) / 2;
    for (int j = threadIdx.x; j < FEAT; j += blockDim.x) {
        float v;
        if (j < i)       v = L[(size_t)k * NTRI + tri_i + j];
        else if (j == i) v = Dd[(size_t)k * FEAT + i];
        else             v = U[(size_t)k * NTRI + j * (j - 1) / 2 + i];
        R[base + j] = f2bf(v);
    }
}

// ---- per-class ball list (ascending n, padded) ----
__global__ void k_classlist(const int* __restrict__ blab, int* __restrict__ list, int* __restrict__ cnt) {
    int k = threadIdx.x;
    if (k >= NUM_LABELS) return;
    int c = 0;
    for (int n = 0; n < NB; n++)
        if (blab[n] == k && c < 16) list[k * 16 + c++] = n;
    cnt[k] = (c > 0) ? c : 1;
    int pad = (c > 0) ? list[k * 16] : 0;
    for (int j = c; j < 16; j++) list[k * 16 + j] = pad;
}

// ---- squared distances, stored n-major: d2T[n][m] ----
__global__ void k_pairwise(const float* __restrict__ pooled, const float* __restrict__ ood,
                           const float* __restrict__ cent,
                           float* __restrict__ d2pT, float* __restrict__ d2oT) {
    __shared__ float q[FEAT];
    int x = blockIdx.x;
    int src = x >> 8, m = x & 255;
    const float* Q = src ? (ood + (size_t)m * FEAT) : (pooled + (size_t)m * FEAT);
    float* OUT = src ? d2oT : d2pT;
    for (int j = threadIdx.x; j < FEAT; j += blockDim.x) q[j] = Q[j];
    __syncthreads();
    int w = threadIdx.x >> 5, lane = threadIdx.x & 31;
    for (int n = w; n < NB; n += 8) {
        const float* C = cent + (size_t)n * FEAT;
        float p = 0.f;
        for (int j = lane; j < FEAT; j += 32) { float d = q[j] - C[j]; p += d * d; }
        for (int off = 16; off; off >>= 1) p += __shfl_down(p, off, 32);
        if (lane == 0) OUT[(size_t)n * BATCH + m] = p;
    }
}

// ---- nearest same-class ball per pooled sample ----
__global__ void k_assign(const float* __restrict__ d2pT, const int* __restrict__ labels,
                         const int* __restrict__ blab, const float* __restrict__ delta,
                         int* __restrict__ ballid, int* __restrict__ clsb, float* __restrict__ dsel) {
    int b = blockIdx.x, lane = threadIdx.x;
    int lab = labels[b];
    float best = 3.4e38f; int bestn = 0x7fffffff;
    for (int n = lane; n < NB; n += 32) {
        if (blab[n] == lab) {
            float v = d2pT[(size_t)n * BATCH + b];
            if (v < best) { best = v; bestn = n; }
        }
    }
    for (int off = 16; off; off >>= 1) {
        float ov = __shfl_down(best, off, 32);
        int   on = __shfl_down(bestn, off, 32);
        if (ov < best || (ov == best && on < bestn)) { best = ov; bestn = on; }
    }
    if (lane == 0) { ballid[b] = bestn; clsb[b] = lab; dsel[b] = delta[bestn]; }
}

// ---- per (class k, ood m): nearest class-k ball (local index into classlist) ----
__global__ void k_nearest(const float* __restrict__ d2oT, const int* __restrict__ list,
                          const int* __restrict__ cnt, const float* __restrict__ delta,
                          int* __restrict__ selj, float* __restrict__ dnear) {
    int k = blockIdx.x, m = threadIdx.x;
    int c = cnt[k];
    float best = 3.4e38f; int bj = 0, bn = list[k * 16];
    for (int j = 0; j < c; j++) {
        int n = list[k * 16 + j];
        float v = d2oT[(size_t)n * BATCH + m];
        if (v < best) { best = v; bj = j; bn = n; }
    }
    selj[k * BATCH + m] = bj;
    dnear[k * BATCH + m] = delta[bn];
}

// ---- Rc[k][i][j] = (R[k] @ centroid(ball j of class k))[i]  via WMMA ----
__global__ void k_rc_gemm(const u16* __restrict__ R, const u16* __restrict__ centb,
                          const int* __restrict__ list, float* __restrict__ Rc) {
    int k = blockIdx.y;
    int w = threadIdx.x >> 5, lane = threadIdx.x & 31;
    int mt = blockIdx.x * 4 + w;
    int hi = lane >> 4, ln = lane & 15;
    int rowA = mt * 16 + ln;
    const u16* ap = R + ((size_t)k * FEAT + rowA) * FEAT + (hi ? 8 : 0);
    int ball = list[k * 16 + ln];
    const u16* bp = centb + (size_t)ball * FEAT + (hi ? 8 : 0);
    v8f acc = {};
#pragma unroll 4
    for (int kk = 0; kk < FEAT; kk += 32) {
        v16bf a  = ldfrag(ap + kk, ap + kk + 16);
        v16bf bb = ldfrag(bp + kk, bp + kk + 16);
        acc = __builtin_amdgcn_wmma_f32_16x16x32_bf16(false, a, false, bb, (short)0, acc, false, false);
    }
    size_t base = ((size_t)k * FEAT + mt * 16) * 16 + ln;
#pragma unroll
    for (int r = 0; r < 8; r++) {
        int i = hi ? (8 + r) : r;
        Rc[base + (size_t)i * 16] = acc[r];
    }
}

// ---- main batched GEMM: a = R[k]@ood[m]; accumulate ||a||^2, a.Rc_sel, ||Rc_sel||^2 ----
// B tile (16 ood columns x K=768, bf16) is DMA'd to LDS once per workgroup in
// fragment order via global_load_async_to_lds_b128: slot (kstep s, lane l) holds
// the exact 32 contiguous bytes lane l consumes at K-step s.
__global__ void k_ood_gemm(const u16* __restrict__ R, const u16* __restrict__ oodb,
                           const float* __restrict__ Rc, const int* __restrict__ selj,
                           const float* __restrict__ dnear, float* __restrict__ contrib) {
    __shared__ __align__(16) u16 ldsB[24 * 32 * 16];   // 24 KB: [s][lane][16 elems]
    __shared__ float part[3][4][32];
    int k = blockIdx.y, nt = blockIdx.x;
    int tid = threadIdx.x;
    int w = tid >> 5, lane = tid & 31;
    int hi = lane >> 4, ln = lane & 15;

    // --- async DMA stage of B tile (1536 x 16B chunks, 12 per thread) ---
    unsigned ldsBase = (unsigned)(size_t)(&ldsB[0]);
#pragma unroll
    for (int i = 0; i < 12; i++) {
        int c = tid + 128 * i;                 // 0..1535
        int h = c & 1, l = (c >> 1) & 31, s = c >> 6;
        const u16* g = oodb + (size_t)(nt * 16 + (l & 15)) * FEAT
                            + s * 32 + (l >> 4) * 8 + h * 16;
        unsigned loff = ldsBase + (unsigned)c * 16u;
        asm volatile("global_load_async_to_lds_b128 %0, %1, off"
                     :: "v"(loff), "v"((unsigned long long)(size_t)g) : "memory");
    }
    asm volatile("s_wait_asynccnt 0x0" ::: "memory");
    __syncthreads();

    int m = nt * 16 + ln;
    int sel = selj[k * BATCH + m];
    const float* rcp = Rc + ((size_t)k * FEAT) * 16 + sel;
    float s1 = 0.f, s2 = 0.f, s3 = 0.f;
    for (int mt = w; mt < 48; mt += 4) {
        int rowA = mt * 16 + ln;
        const u16* ap = R + ((size_t)k * FEAT + rowA) * FEAT + (hi ? 8 : 0);
        if (mt + 4 < 48) __builtin_prefetch(R + ((size_t)k * FEAT + rowA + 64) * FEAT, 0, 1);
        v8f acc = {};
#pragma unroll 4
        for (int kk = 0, s = 0; kk < FEAT; kk += 32, s++) {
            const u16* lb = &ldsB[(size_t)(s * 32 + lane) * 16];
            v16bf a  = ldfrag(ap + kk, ap + kk + 16);
            v16bf bb = ldfrag(lb, lb + 8);
            acc = __builtin_amdgcn_wmma_f32_16x16x32_bf16(false, a, false, bb, (short)0, acc, false, false);
        }
        int ibase = mt * 16 + (hi ? 8 : 0);
#pragma unroll
        for (int r = 0; r < 8; r++) {
            float cv = acc[r];
            float rc = rcp[(size_t)(ibase + r) * 16];
            s1 += cv * cv; s2 += cv * rc; s3 += rc * rc;
        }
    }
    part[0][w][lane] = s1; part[1][w][lane] = s2; part[2][w][lane] = s3;
    __syncthreads();
    if (tid < 16) {
        int col = tid;
        float t1 = 0.f, t2 = 0.f, t3 = 0.f;
        for (int ww = 0; ww < 4; ww++) {
            t1 += part[0][ww][col] + part[0][ww][col + 16];
            t2 += part[1][ww][col] + part[1][ww][col + 16];
            t3 += part[2][ww][col] + part[2][ww][col + 16];
        }
        float e2  = t1 - 2.f * t2 + t3;
        float euc = sqrtf(fmaxf(e2, 0.f));
        int mm = nt * 16 + col;
        float dn = dnear[k * BATCH + mm];
        float diff = dn - euc;
        contrib[(size_t)k * BATCH + mm] = (dn > euc) ? (diff + 0.1f) : (0.1f * expf(diff));
    }
}

// ---- pos branch in full f32, R entries generated on the fly from L/U/Dd ----
__global__ void k_posloss(const float* __restrict__ pooled, const float* __restrict__ cent,
                          const float* __restrict__ L, const float* __restrict__ U,
                          const float* __restrict__ Dd, const int* __restrict__ ballid,
                          const int* __restrict__ clsb, const float* __restrict__ dselv,
                          float* __restrict__ posloss, float* __restrict__ gtb, float* __restrict__ ltb) {
    __shared__ float xd[FEAT];
    __shared__ float red[256];
    int b = blockIdx.x, t = threadIdx.x;
    int ball = ballid[b], cls = clsb[b];
    for (int j = t; j < FEAT; j += 256)
        xd[j] = pooled[(size_t)b * FEAT + j] - cent[(size_t)ball * FEAT + j];
    __syncthreads();
    const float* Lk = L + (size_t)cls * NTRI;
    const float* Uk = U + (size_t)cls * NTRI;
    const float* Dk = Dd + (size_t)cls * FEAT;
    float ssq = 0.f;
    for (int r = 0; r < 3; r++) {
        int i = t * 3 + r;
        int tri_i = i * (i - 1) / 2;
        float acc = Dk[i] * xd[i];
        for (int j = 0; j < i; j++) acc += Lk[tri_i + j] * xd[j];
        int tj = (i + 1) * i / 2 + i;           // U index for j=i+1, then tj += j
        for (int j = i + 1; j < FEAT; j++) { acc += Uk[tj] * xd[j]; tj += j; }
        ssq += acc * acc;
    }
    red[t] = ssq; __syncthreads();
    for (int s = 128; s; s >>= 1) { if (t < s) red[t] += red[t + s]; __syncthreads(); }
    if (t == 0) {
        float euc = sqrtf(red[0]);
        float d = dselv[b];
        posloss[b] = (d > euc) ? expf(euc - d) : (euc - d);
        gtb[b] = (euc > d) ? 1.f : 0.f;
        ltb[b] = (euc < d) ? 1.f : 0.f;
    }
}

// ---- final deterministic reduction -> 5 scalars ----
__global__ void k_final(const float* __restrict__ posloss, const float* __restrict__ gtb,
                        const float* __restrict__ ltb, const float* __restrict__ contrib,
                        float* __restrict__ out) {
    __shared__ float r0[256], r1[256], r2[256], r3[256];
    int t = threadIdx.x;
    float s = 0.f;
    for (int k = 0; k < NUM_LABELS; k++) s += contrib[(size_t)k * BATCH + t];
    r0[t] = posloss[t]; r1[t] = gtb[t]; r2[t] = ltb[t]; r3[t] = s;
    __syncthreads();
    for (int st = 128; st; st >>= 1) {
        if (t < st) { r0[t] += r0[t + st]; r1[t] += r1[t + st]; r2[t] += r2[t + st]; r3[t] += r3[t + st]; }
        __syncthreads();
    }
    if (t == 0) {
        float pm = r0[0] / 256.f, nm = r3[0] / 256.f;
        out[0] = pm; out[1] = nm; out[2] = r1[0]; out[3] = r2[0]; out[4] = pm + nm;
    }
}

extern "C" void kernel_launch(void* const* d_in, const int* in_sizes, int n_in,
                              void* d_out, int out_size, void* d_ws, size_t ws_size,
                              hipStream_t stream) {
    const float* pooled = (const float*)d_in[0];
    const float* ood    = (const float*)d_in[1];
    const float* cent   = (const float*)d_in[2];
    const float* delta  = (const float*)d_in[3];
    const float* L      = (const float*)d_in[4];
    const float* U      = (const float*)d_in[5];
    const float* Dd     = (const float*)d_in[6];
    const int*   labels = (const int*)d_in[7];
    const int*   blab   = (const int*)d_in[8];
    float* out = (float*)d_out;

    char* ws = (char*)d_ws;
    size_t off = 0;
    auto take = [&](size_t bytes) -> void* {
        void* p = ws + off; off = (off + bytes + 255) & ~(size_t)255; return p;
    };
    u16*   Rb     = (u16*)take((size_t)NUM_LABELS * FEAT * FEAT * 2);
    u16*   centb  = (u16*)take((size_t)NB * FEAT * 2);
    u16*   oodb   = (u16*)take((size_t)BATCH * FEAT * 2);
    float* d2pT   = (float*)take((size_t)NB * BATCH * 4);
    float* d2oT   = (float*)take((size_t)NB * BATCH * 4);
    float* Rc     = (float*)take((size_t)NUM_LABELS * FEAT * 16 * 4);
    int*   list   = (int*)take((size_t)NUM_LABELS * 16 * 4);
    int*   cnt    = (int*)take((size_t)NUM_LABELS * 4);
    int*   selj   = (int*)take((size_t)NUM_LABELS * BATCH * 4);
    float* dnear  = (float*)take((size_t)NUM_LABELS * BATCH * 4);
    int*   ballid = (int*)take((size_t)BATCH * 4);
    int*   clsb   = (int*)take((size_t)BATCH * 4);
    float* dsel   = (float*)take((size_t)BATCH * 4);
    float* ploss  = (float*)take((size_t)BATCH * 4);
    float* gtb    = (float*)take((size_t)BATCH * 4);
    float* ltb    = (float*)take((size_t)BATCH * 4);
    float* contrib= (float*)take((size_t)NUM_LABELS * BATCH * 4);

    k_convert  <<<NB + BATCH, 256, 0, stream>>>(cent, ood, centb, oodb);
    k_build_R  <<<NUM_LABELS * FEAT, 256, 0, stream>>>(L, U, Dd, Rb);
    k_classlist<<<1, 256, 0, stream>>>(blab, list, cnt);
    k_pairwise <<<512, 256, 0, stream>>>(pooled, ood, cent, d2pT, d2oT);
    k_assign   <<<BATCH, 32, 0, stream>>>(d2pT, labels, blab, delta, ballid, clsb, dsel);
    k_nearest  <<<NUM_LABELS, BATCH, 0, stream>>>(d2oT, list, cnt, delta, selj, dnear);
    k_rc_gemm  <<<dim3(12, NUM_LABELS), 128, 0, stream>>>(Rb, centb, list, Rc);
    k_ood_gemm <<<dim3(16, NUM_LABELS), 128, 0, stream>>>(Rb, oodb, Rc, selj, dnear, contrib);
    k_posloss  <<<BATCH, 256, 0, stream>>>(pooled, cent, L, U, Dd, ballid, clsb, dsel, ploss, gtb, ltb);
    k_final    <<<1, 256, 0, stream>>>(ploss, gtb, ltb, contrib, out);
}
